// MultiHeadedAttention_82944408420459
// MI455X (gfx1250) — matrix operand order repaired and checked by reading, not compiled, not run on hardware
//
#include <hip/hip_runtime.h>
#include <hip/hip_bf16.h>

// ---------------------------------------------------------------------------
// MultiHeadedAttention on MI455X (gfx1250): bf16 WMMA + fused softmax in LDS
// + async global->LDS staging of V (ASYNCcnt overlap), NT hints for streams,
// one-shot f32->bf16 conversion pass so GEMM inner loops are conversion-free
// ---------------------------------------------------------------------------

typedef __bf16 bf16;
typedef __attribute__((ext_vector_type(16))) __bf16 v16bf;
typedef __attribute__((ext_vector_type(4)))  __bf16 v4bf;
typedef __attribute__((ext_vector_type(8)))  float  v8f;
typedef __attribute__((ext_vector_type(4)))  float  v4f;

#define B_DIM 8
#define S_DIM 1024
#define E_DIM 1024
#define H_DIM 16
#define HD_DIM 64
#define M_DIM (B_DIM * S_DIM)      // 8192
#define LROW 1025                  // padded f32 LDS row stride -> conflict-free
#define ABROW 1032                 // padded bf16 LDS row stride -> conflict-free

__device__ __forceinline__ int lane_id() { return threadIdx.x & 31; }

// f32 -> bf16: native conversion (v_cvt_pk_bf16_f32 when paired)
__device__ __forceinline__ bf16 f2bf(float f) { return (bf16)f; }

__device__ __forceinline__ v8f wmma_bf16(v16bf a, v16bf b, v8f c) {
    return __builtin_amdgcn_wmma_f32_16x16x32_bf16(
        /*neg_a=*/false, a, /*neg_b=*/false, b,
        /*c_mod=*/(short)0, c, /*reuse_a=*/false, /*reuse_b=*/false);
}

// A fragment (16x32, MxK): src points at tile origin [m0,k0], row-major, ld = row stride.
// lane l: M = l&15 ; K-base = (l>=16 ? 16 : 0); VGPR j: K = base + 2j
__device__ __forceinline__ v16bf load_a_bf16(const bf16* src, int ld) {
    const int l = lane_id();
    const bf16* row = src + (l & 15) * ld + ((l >> 4) << 4);
    v16bf a;
#pragma unroll
    for (int j = 0; j < 8; ++j) {
        const int k = 2 * j;
        a[2 * j]     = row[k];
        a[2 * j + 1] = row[k + 1];
    }
    return a;
}

// B fragment (32x16, KxN) from row-major [N,K] data (K contiguous): B[k][n]=src[n*ld+k].
// lane l: N = l&15 ; K-base = (l>=16 ? 16 : 0); VGPR j: K = base + 2j
__device__ __forceinline__ v16bf load_b_nk_bf16(const bf16* src, int ld) {
    const int l = lane_id();
    const bf16* row = src + (l & 15) * ld + ((l >> 4) << 4);
    v16bf b;
#pragma unroll
    for (int j = 0; j < 8; ++j) {
        b[2 * j]     = row[2 * j];
        b[2 * j + 1] = row[2 * j + 1];
    }
    return b;
}

// B fragment (32x16, KxN) from row-major [K,N] data (N contiguous): B[k][n]=src[k*ld+n].
__device__ __forceinline__ v16bf load_b_kn_bf16(const bf16* src, int ld) {
    const int l = lane_id();
    const bf16* col = src + (l & 15) + (((l >> 4) << 4) * ld);
    v16bf b;
#pragma unroll
    for (int j = 0; j < 8; ++j) {
        b[2 * j]     = col[(2 * j) * ld];
        b[2 * j + 1] = col[(2 * j + 1) * ld];
    }
    return b;
}

// ---------------------------------------------------------------------------
// Kernel 0: streaming f32 -> bf16 convert (NT loads: sources are read-once)
// ---------------------------------------------------------------------------
__global__ void __launch_bounds__(256)
cvt_kernel(const float* __restrict__ src, bf16* __restrict__ dst, int n) {
    const int i = (blockIdx.x * 256 + threadIdx.x) * 4;
    if (i < n) {
        const v4f f = __builtin_nontemporal_load((const v4f*)(src + i));
        v4bf o;
        o[0] = f2bf(f[0]); o[1] = f2bf(f[1]); o[2] = f2bf(f[2]); o[3] = f2bf(f[3]);
        *(v4bf*)(dst + i) = o;
    }
}

// ---------------------------------------------------------------------------
// Kernel 1: Y[M,E] = X[M,E] @ W^T + bias   (all-bf16 operands, bf16 out)
// block = 128 threads (4 waves); wave -> 32 rows x 64 cols (B frags reused 2x)
// grid = (M/128, E/64)
// ---------------------------------------------------------------------------
__global__ void __launch_bounds__(128)
proj_kernel(const bf16* __restrict__ X, const bf16* __restrict__ W,
            const float* __restrict__ bias, bf16* __restrict__ Y) {
    const int wave = threadIdx.x >> 5;
    const int l = lane_id();
    const int m0 = (blockIdx.x * 4 + wave) * 32;
    const int n0 = blockIdx.y * 64;

    v8f c[2][4] = {};
    for (int k0 = 0; k0 < E_DIM; k0 += 32) {
        v16bf a0 = load_a_bf16(X + (size_t)m0 * E_DIM + k0, E_DIM);
        v16bf a1 = load_a_bf16(X + (size_t)(m0 + 16) * E_DIM + k0, E_DIM);
#pragma unroll
        for (int f = 0; f < 4; ++f) {
            v16bf b = load_b_nk_bf16(W + (size_t)(n0 + f * 16) * E_DIM + k0, E_DIM);
            c[0][f] = wmma_bf16(a0, b, c[0][f]);
            c[1][f] = wmma_bf16(a1, b, c[1][f]);
        }
    }
#pragma unroll
    for (int t = 0; t < 2; ++t)
#pragma unroll
        for (int f = 0; f < 4; ++f)
#pragma unroll
            for (int r = 0; r < 8; ++r) {
                const int row = m0 + t * 16 + r + ((l >> 4) << 3);
                const int col = n0 + f * 16 + (l & 15);
                Y[(size_t)row * E_DIM + col] = f2bf(c[t][f][r] + bias[col]);
            }
}

// ---------------------------------------------------------------------------
// Kernel 2 (fused): per (b,h,16-row tile):
//   issue async global->LDS copies of this wave's 64x64 V sub-tile  (ASYNCcnt)
//   logits = Q K^T * 0.125 - 1e9*mask   -> LDS (16 x 1024 f32, padded rows)
//   softmax rows (wave32 shuffles), + adjoin; NT-store attn (f32) once to HBM
//   and keep a bf16 copy in LDS (A-fragment source, conversion-free phase C)
//   s_wait_asynccnt 0 ; x = attn @ V(LDS)  (K split over 16 waves, ds_add_f32)
// block = 512 threads (16 waves), grid = (S/16, B*H)
// LDS: logits 65,600 + acc 4,096 + vstage 131,072 + abf 33,024 = 233,792 B
// ---------------------------------------------------------------------------
__global__ void __launch_bounds__(512)
attn_kernel(const bf16* __restrict__ Q, const bf16* __restrict__ Kmat,
            const bf16* __restrict__ V, const int* __restrict__ mask,
            const float* __restrict__ adjoin, float* __restrict__ attn_out,
            bf16* __restrict__ Xout) {
    extern __shared__ char smem_raw[];
    float* logits = (float*)smem_raw;                    // 16 * LROW f32
    float* acc    = logits + 16 * LROW;                  // 16 * 64  f32
    bf16*  vstage = (bf16*)(acc + 16 * 64);              // 1024 * 64 bf16
    bf16*  abf    = vstage + S_DIM * HD_DIM;             // 16 * ABROW bf16

    const int bh = blockIdx.y;            // 0..127
    const int b = bh >> 4, h = bh & 15;
    const int s0 = blockIdx.x * 16;
    const int wave = threadIdx.x >> 5;
    const int l = lane_id();
    const int tid = threadIdx.x;

    // ---- async stage: wave w copies V rows [w*64, w*64+64) x 64 cols to LDS ----
    // 64 rows x 128 B = 8 KB per wave; 16 B/lane/op, 16 ops/lane; overlaps phase A/B
    {
        const bf16* vsrc = V + ((size_t)(b * S_DIM + wave * 64)) * E_DIM + h * HD_DIM;
        const unsigned vbase = (unsigned)(uintptr_t)(vstage + wave * 64 * 64);
#pragma unroll
        for (int i = 0; i < 16; ++i) {
            const int t   = i * 32 + l;     // 0..511
            const int row = t >> 3;         // 0..63
            const int seg = t & 7;          // 16-byte segment in the 128 B row
            const bf16* g = vsrc + (size_t)row * E_DIM + seg * 8;
            const unsigned ldsoff = vbase + (unsigned)((row * 64 + seg * 8) * 2);
            asm volatile("global_load_async_to_lds_b128 %0, %1, off"
                         :: "v"(ldsoff), "v"(g) : "memory");
        }
    }

    // zero the cross-wave accumulator
    for (int i = tid; i < 16 * 64; i += 512) acc[i] = 0.0f;

    // ---- Phase A: logits strip, cols [wave*64, wave*64+64) ----
    const int j0 = wave * 64;
    const bf16* qb = Q + ((size_t)(b * S_DIM + s0)) * E_DIM + h * HD_DIM;
    v8f c[4] = {};
#pragma unroll
    for (int d0 = 0; d0 < HD_DIM; d0 += 32) {
        v16bf a = load_a_bf16(qb + d0, E_DIM);
#pragma unroll
        for (int f = 0; f < 4; ++f) {
            const bf16* kb =
                Kmat + ((size_t)(b * S_DIM + j0 + f * 16)) * E_DIM + h * HD_DIM + d0;
            v16bf bb = load_b_nk_bf16(kb, E_DIM);
            c[f] = wmma_bf16(a, bb, c[f]);
        }
    }
    const int* mrow = mask + ((size_t)b * S_DIM + s0) * S_DIM;
#pragma unroll
    for (int f = 0; f < 4; ++f)
#pragma unroll
        for (int r = 0; r < 8; ++r) {
            const int rr = r + ((l >> 4) << 3);
            const int cc = j0 + f * 16 + (l & 15);
            const float m = (float)__builtin_nontemporal_load(&mrow[(size_t)rr * S_DIM + cc]);
            logits[rr * LROW + cc] = c[f][r] * 0.125f + m * -1e9f;
        }
    __syncthreads();

    // ---- Phase B: softmax on row 'wave', + adjoin; f32 attn to HBM (NT),
    //      bf16 attn to LDS for phase C ----
    {
        const int row = wave;
        float vals[32];
        float mx = -3.402823466e+38f;
#pragma unroll
        for (int i = 0; i < 32; ++i) {
            vals[i] = logits[row * LROW + l + i * 32];
            mx = fmaxf(mx, vals[i]);
        }
#pragma unroll
        for (int off = 16; off > 0; off >>= 1) mx = fmaxf(mx, __shfl_xor(mx, off, 32));
        float sum = 0.0f;
#pragma unroll
        for (int i = 0; i < 32; ++i) {
            vals[i] = __expf(vals[i] - mx);
            sum += vals[i];
        }
#pragma unroll
        for (int off = 16; off > 0; off >>= 1) sum += __shfl_xor(sum, off, 32);
        const float inv = 1.0f / sum;
        const float* adjr = adjoin + ((size_t)b * S_DIM + s0 + row) * S_DIM;
        float* arow = attn_out + ((size_t)bh * S_DIM + s0 + row) * S_DIM;
#pragma unroll
        for (int i = 0; i < 32; ++i) {
            const int cc = l + i * 32;
            const float av = vals[i] * inv + __builtin_nontemporal_load(&adjr[cc]);
            abf[row * ABROW + cc] = f2bf(av);            // phase-C A source
            __builtin_nontemporal_store(av, &arow[cc]);  // attn: write-once stream
        }
    }
    __syncthreads();

    // ---- Phase C: x = attn @ V(LDS) ; wave owns K range [wave*64, +64) ----
    asm volatile("s_wait_asynccnt 0x0" ::: "memory");  // this wave's V tile is in LDS

    v8f xc[4] = {};
    const int kk0 = wave * 64;
#pragma unroll
    for (int kq = 0; kq < 2; ++kq) {
        const int kk = kk0 + kq * 32;
        v16bf a = load_a_bf16(abf + kk, ABROW);          // bf16 straight from LDS
#pragma unroll
        for (int f = 0; f < 4; ++f) {
            const bf16* vb = vstage + (size_t)kk * HD_DIM + f * 16;  // LDS, ld=64
            v16bf bb = load_b_kn_bf16(vb, HD_DIM);
            xc[f] = wmma_bf16(a, bb, xc[f]);
        }
    }
    // reduce partial products across waves with LDS float atomics (ds_add_f32)
#pragma unroll
    for (int f = 0; f < 4; ++f)
#pragma unroll
        for (int r = 0; r < 8; ++r) {
            const int rr = r + ((l >> 4) << 3);
            const int cc = f * 16 + (l & 15);
            atomicAdd(&acc[rr * 64 + cc], xc[f][r]);
        }
    __syncthreads();

    bf16* xout = Xout + ((size_t)(b * S_DIM + s0)) * E_DIM + h * HD_DIM;
    for (int i = tid; i < 16 * 64; i += 512) {
        const int rr = i >> 6, cc = i & 63;
        xout[(size_t)rr * E_DIM + cc] = f2bf(acc[rr * 64 + cc]);
    }
}

// ---------------------------------------------------------------------------
// Kernel 3: out[M,E] = X[M,E](bf16) @ Wo^T(bf16) + bo  (f32 out)
// ---------------------------------------------------------------------------
__global__ void __launch_bounds__(128)
out_proj_kernel(const bf16* __restrict__ X, const bf16* __restrict__ W,
                const float* __restrict__ bias, float* __restrict__ Y) {
    const int wave = threadIdx.x >> 5;
    const int l = lane_id();
    const int m0 = (blockIdx.x * 4 + wave) * 32;
    const int n0 = blockIdx.y * 64;

    v8f c[2][4] = {};
    for (int k0 = 0; k0 < E_DIM; k0 += 32) {
        v16bf a0 = load_a_bf16(X + (size_t)m0 * E_DIM + k0, E_DIM);
        v16bf a1 = load_a_bf16(X + (size_t)(m0 + 16) * E_DIM + k0, E_DIM);
#pragma unroll
        for (int f = 0; f < 4; ++f) {
            v16bf b = load_b_nk_bf16(W + (size_t)(n0 + f * 16) * E_DIM + k0, E_DIM);
            c[0][f] = wmma_bf16(a0, b, c[0][f]);
            c[1][f] = wmma_bf16(a1, b, c[1][f]);
        }
    }
#pragma unroll
    for (int t = 0; t < 2; ++t)
#pragma unroll
        for (int f = 0; f < 4; ++f)
#pragma unroll
            for (int r = 0; r < 8; ++r) {
                const int row = m0 + t * 16 + r + ((l >> 4) << 3);
                const int col = n0 + f * 16 + (l & 15);
                Y[(size_t)row * E_DIM + col] = c[t][f][r] + bias[col];
            }
}

// ---------------------------------------------------------------------------
extern "C" void kernel_launch(void* const* d_in, const int* in_sizes, int n_in,
                              void* d_out, int out_size, void* d_ws, size_t ws_size,
                              hipStream_t stream) {
    const float* q      = (const float*)d_in[0];
    const float* k      = (const float*)d_in[1];
    const float* v      = (const float*)d_in[2];
    const int*   mask   = (const int*)d_in[3];
    const float* adjoin = (const float*)d_in[4];
    const float* Wq = (const float*)d_in[5];
    const float* bq = (const float*)d_in[6];
    const float* Wk = (const float*)d_in[7];
    const float* bk = (const float*)d_in[8];
    const float* Wv = (const float*)d_in[9];
    const float* bv = (const float*)d_in[10];
    const float* Wo = (const float*)d_in[11];
    const float* bo = (const float*)d_in[12];

    float* out  = (float*)d_out;
    float* attn = out + (size_t)M_DIM * E_DIM;   // outputs: (out, attn) flat

    const size_t NME = (size_t)M_DIM * E_DIM;    // 8,388,608
    const size_t NEE = (size_t)E_DIM * E_DIM;    // 1,048,576
    bf16* xq  = (bf16*)d_ws;          // bf16 copies of inputs
    bf16* xk  = xq  + NME;
    bf16* xv  = xk  + NME;
    bf16* cWq = xv  + NME;            // bf16 copies of weights
    bf16* cWk = cWq + NEE;
    bf16* cWv = cWk + NEE;
    bf16* cWo = cWv + NEE;
    bf16* pq  = cWo + NEE;            // projected Q/K/V (split-head addressing)
    bf16* pk  = pq  + NME;
    bf16* pv  = pk  + NME;
    bf16* px  = pv  + NME;            // attention context (pre output proj)

    // one-shot f32 -> bf16 conversions
    {
        const int nb_me = (int)(NME / 4 / 256);   // 8192 blocks
        const int nb_ee = (int)(NEE / 4 / 256);   // 1024 blocks
        cvt_kernel<<<nb_me, 256, 0, stream>>>(q,  xq,  (int)NME);
        cvt_kernel<<<nb_me, 256, 0, stream>>>(k,  xk,  (int)NME);
        cvt_kernel<<<nb_me, 256, 0, stream>>>(v,  xv,  (int)NME);
        cvt_kernel<<<nb_ee, 256, 0, stream>>>(Wq, cWq, (int)NEE);
        cvt_kernel<<<nb_ee, 256, 0, stream>>>(Wk, cWk, (int)NEE);
        cvt_kernel<<<nb_ee, 256, 0, stream>>>(Wv, cWv, (int)NEE);
        cvt_kernel<<<nb_ee, 256, 0, stream>>>(Wo, cWo, (int)NEE);
    }

    const dim3 gemm_grid(M_DIM / 128, E_DIM / 64);   // (64, 16)
    const dim3 gemm_block(128);

    proj_kernel<<<gemm_grid, gemm_block, 0, stream>>>(xq, cWq, bq, pq);
    proj_kernel<<<gemm_grid, gemm_block, 0, stream>>>(xk, cWk, bk, pk);
    proj_kernel<<<gemm_grid, gemm_block, 0, stream>>>(xv, cWv, bv, pv);

    const dim3 attn_grid(S_DIM / 16, B_DIM * H_DIM);  // (64, 128)
    const size_t smem = (size_t)(16 * LROW + 16 * 64) * sizeof(float)
                      + (size_t)S_DIM * HD_DIM * 2        // vstage
                      + (size_t)16 * ABROW * 2;           // abf
    attn_kernel<<<attn_grid, 512, smem, stream>>>(pq, pk, pv, mask, adjoin,
                                                  attn, px);

    out_proj_kernel<<<gemm_grid, gemm_block, 0, stream>>>(px, cWo, bo, out);
}
